// SynthesisLayer_36507222016808
// MI455X (gfx1250) — matrix operand order repaired and checked by reading, not compiled, not run on hardware
//
#include <hip/hip_runtime.h>

typedef __attribute__((ext_vector_type(16))) __bf16       v16bf;
typedef __attribute__((ext_vector_type(8)))  float        v8f;
typedef __attribute__((ext_vector_type(4)))  unsigned int u32x4;
typedef __attribute__((ext_vector_type(4)))  int          v4i;

union Frag {
  v16bf v;
  u32x4 u[2];
};

#define B_    16
#define CIN_  512
#define COUT_ 512
#define RES_  64
#define PX_   (RES_ * RES_)
#define WDIM_ 512

// gfx1250 async global->LDS path (ASYNCcnt tracked). Guarded so the file
// compiles on toolchains lacking the builtins (falls back to sync staging).
// Parameter types per clang-22 diagnostic: (v4i AS1*, v4i AS3*, Imm int, Imm int)
#if __has_builtin(__builtin_amdgcn_global_load_async_to_lds_b128) && \
    __has_builtin(__builtin_amdgcn_s_wait_asynccnt)
#define USE_ASYNC_LDS 1
#define ASYNC_B128(gp, lp)                                              \
  __builtin_amdgcn_global_load_async_to_lds_b128(                       \
      (__attribute__((address_space(1))) v4i*)(gp),                     \
      (__attribute__((address_space(3))) v4i*)(lp), 0, 0)
#else
#define USE_ASYNC_LDS 0
#endif

__device__ __forceinline__ unsigned short f2bf(float f) {
  union { float f; unsigned int u; } c; c.f = f;
  unsigned int u = c.u;
  u += 0x7FFFu + ((u >> 16) & 1u);   // round-to-nearest-even
  return (unsigned short)(u >> 16);
}

// ---------------------------------------------------------------------------
// K1: styles[b][c] = m1*m2 + m3, proj = w @ (affine_w/sqrt(wdim)).T + affine_b
// ---------------------------------------------------------------------------
__global__ void k_styles(const float* __restrict__ w,
                         const float* __restrict__ aw,
                         const float* __restrict__ ab,
                         float* __restrict__ styles) {
  int id = blockIdx.x * blockDim.x + threadIdx.x;
  if (id >= B_ * CIN_) return;
  int b = id / CIN_, c = id % CIN_;
  const float inv = 0.04419417382415922f; // 1/sqrt(512)
  float s1 = 0.f, s2 = 0.f, s3 = 0.f;
  const float* wr = w + (size_t)b * WDIM_;
  const float* a1 = aw + (size_t)c * WDIM_;
  const float* a2 = aw + (size_t)(CIN_ + c) * WDIM_;
  const float* a3 = aw + (size_t)(2 * CIN_ + c) * WDIM_;
  for (int d = 0; d < WDIM_; ++d) {
    float wv = wr[d];
    s1 += wv * a1[d];
    s2 += wv * a2[d];
    s3 += wv * a3[d];
  }
  float m1 = s1 * inv + ab[c];
  float m2 = s2 * inv + ab[CIN_ + c];
  float m3 = s3 * inv + ab[2 * CIN_ + c];
  styles[id] = m1 * m2 + m3;
}

// ---------------------------------------------------------------------------
// K2a: wsq[co][ci] = sum_k weight[co][ci][k]^2
// ---------------------------------------------------------------------------
__global__ void k_wsq(const float* __restrict__ weight, float* __restrict__ wsq) {
  int id = blockIdx.x * blockDim.x + threadIdx.x;
  if (id >= COUT_ * CIN_) return;
  const float* p = weight + (size_t)id * 9;
  float s = 0.f;
  #pragma unroll
  for (int i = 0; i < 9; ++i) s += p[i] * p[i];
  wsq[id] = s;
}

// ---------------------------------------------------------------------------
// K2b: dcoefs[b][co] = rsqrt( sum_ci wsq[co][ci]*styles[b][ci]^2 + 1e-8 )
// ---------------------------------------------------------------------------
__global__ void k_dcoef(const float* __restrict__ wsq,
                        const float* __restrict__ styles,
                        float* __restrict__ dcoefs) {
  int id = blockIdx.x * blockDim.x + threadIdx.x;
  if (id >= B_ * COUT_) return;
  int b = id / COUT_, co = id % COUT_;
  const float* wr = wsq + (size_t)co * CIN_;
  const float* sr = styles + (size_t)b * CIN_;
  float s = 0.f;
  for (int ci = 0; ci < CIN_; ++ci) {
    float st = sr[ci];
    s += wr[ci] * st * st;
  }
  dcoefs[id] = rsqrtf(s + 1e-8f);
}

// ---------------------------------------------------------------------------
// K3: wb[t][co][ci] (bf16) from weight[co][ci][t] (f32), t = kh*3+kw
// ---------------------------------------------------------------------------
__global__ void k_packw(const float* __restrict__ weight,
                        unsigned short* __restrict__ wb) {
  int id = blockIdx.x * blockDim.x + threadIdx.x;
  if (id >= 9 * COUT_ * CIN_) return;
  int t  = id / (COUT_ * CIN_);
  int r  = id % (COUT_ * CIN_);
  int co = r / CIN_;
  int ci = r % CIN_;
  wb[id] = f2bf(weight[((size_t)co * CIN_ + ci) * 9 + t]);
}

// ---------------------------------------------------------------------------
// K4: xs[b][px][ci] (bf16) = x[b][ci][px] * styles[b][ci]  (16x16 LDS transpose)
// ---------------------------------------------------------------------------
__global__ void k_packx(const float* __restrict__ x,
                        const float* __restrict__ styles,
                        unsigned short* __restrict__ xs) {
  __shared__ float tile[16][17];
  int b  = blockIdx.z;
  int ct = blockIdx.y;   // ci tile
  int pt = blockIdx.x;   // px tile
  int tx = threadIdx.x & 15;
  int ty = threadIdx.x >> 4;
  int ci = ct * 16 + ty;
  int px = pt * 16 + tx;
  tile[ty][tx] = x[((size_t)b * CIN_ + ci) * PX_ + px] * styles[b * CIN_ + ci];
  __syncthreads();
  int ci2 = ct * 16 + tx;
  int px2 = pt * 16 + ty;
  xs[((size_t)b * PX_ + px2) * CIN_ + ci2] = f2bf(tile[tx][ty]);
}

// ---------------------------------------------------------------------------
// Main: per-batch implicit-GEMM conv3x3 via v_wmma_f32_16x16x32_bf16.
// Block = 256 threads (8 wave32), 64(co) x 256(px = 4 image rows) tile.
// Double-buffered LDS, async global->LDS staging overlapped with WMMA.
// ---------------------------------------------------------------------------
__global__ __launch_bounds__(256) void k_modconv(
    const unsigned short* __restrict__ wb,     // [9][512co][512ci] bf16
    const unsigned short* __restrict__ xs,     // [B][4096px][512ci] bf16
    const float* __restrict__ dcoefs,          // [B][512co]
    const float* __restrict__ bias,            // [512]
    const float* __restrict__ noise,           // [4096]
    const float* __restrict__ nstr,            // [1]
    float* __restrict__ out)                   // [B][512co][4096px]
{
  constexpr int WH = 9 * 64 * 32;   // 18432 halfs / buffer (36 KB)
  constexpr int XH = 6 * 66 * 32;   // 12672 halfs / buffer (24.75 KB)
  __shared__ __align__(16) unsigned short lds_w[2 * WH];   //  72 KB
  __shared__ __align__(16) unsigned short lds_x[2 * XH];   //  49.5 KB

  const int b      = blockIdx.z;
  const int coBase = blockIdx.y * 64;
  const int pxBase = blockIdx.x * 256;   // 4 image rows
  const int y0     = pxBase >> 6;        // first image row of this tile
  const int tid    = threadIdx.x;
  const int lane   = tid & 31;
  const int wave   = tid >> 5;
  const int wm     = wave & 1;           // co direction  (2 waves of 32)
  const int wn     = wave >> 1;          // px direction  (4 waves of 64)
  const int ln16   = lane & 15;
  const int lhalf  = lane >> 4;          // 0/1

  v8f acc[2][4] = {};

  const unsigned short* xb = xs + (size_t)b * PX_ * CIN_;

  // ---- stage one ci-chunk into LDS buffer `buf` ----
  auto issueW = [&](int cc, int buf) {
    unsigned short* lw = lds_w + buf * WH;
    for (int i = tid; i < 2304; i += 256) {        // 9 taps * 64 rows * 4
      int t  = i >> 8;                             // 256 b128 per tap
      int r  = (i >> 2) & 63;
      int c4 = i & 3;
      const unsigned short* gp =
          wb + (((size_t)t * COUT_ + coBase + r) * CIN_ + cc * 32 + c4 * 8);
      unsigned short* lp = lw + ((t * 64 + r) * 32 + c4 * 8);
#if USE_ASYNC_LDS
      ASYNC_B128(gp, lp);
#else
      *(u32x4*)lp = *(const u32x4*)gp;
#endif
    }
  };
  auto issueX = [&](int cc, int buf) {
    unsigned short* lx = lds_x + buf * XH;
    for (int i = tid; i < 1584; i += 256) {        // 6 rows * 66 cols * 4
      int pos = i >> 2;
      int c4  = i & 3;
      int ry  = pos / 66;
      int cx  = pos - ry * 66;
      int gy  = y0 + ry - 1;
      int gx  = cx - 1;
      bool in = (gy >= 0 && gy < RES_ && gx >= 0 && gx < RES_);
#if USE_ASYNC_LDS
      // OOB slots pre-zeroed once in prologue; skip them here.
      if (in)
        ASYNC_B128(xb + ((size_t)(gy * RES_ + gx) * CIN_ + cc * 32 + c4 * 8),
                   lx + (pos * 32 + c4 * 8));
#else
      u32x4 v = {0u, 0u, 0u, 0u};
      if (in)
        v = *(const u32x4*)(xb + ((size_t)(gy * RES_ + gx) * CIN_ + cc * 32 + c4 * 8));
      *(u32x4*)(lx + (pos * 32 + c4 * 8)) = v;
#endif
    }
  };

  // ---- 9 taps x 8 WMMAs per wave from LDS buffer `buf` ----
  auto computeChunk = [&](int buf) {
    const unsigned short* lw = lds_w + buf * WH;
    const unsigned short* lx = lds_x + buf * XH;
    #pragma unroll
    for (int t = 0; t < 9; ++t) {
      const int dy = t / 3;
      const int dx = t - dy * 3;
      // A fragments (W rows): lane holds row M=ln16; K-split halves at +0/+16
      Frag a0, a1;
      int row0  = wm * 32 + ln16;
      int base0 = (t * 64 + row0) * 32 + lhalf * 8;
      a0.u[0] = *(const u32x4*)(lw + base0);
      a0.u[1] = *(const u32x4*)(lw + base0 + 16);
      a1.u[0] = *(const u32x4*)(lw + base0 + 16 * 32);
      a1.u[1] = *(const u32x4*)(lw + base0 + 16 * 32 + 16);
      // All 4 B fragments first, then the 8 WMMAs (more ILP for scheduler).
      Frag bf[4];
      #pragma unroll
      for (int ni = 0; ni < 4; ++ni) {
        int pxl  = wn * 64 + ni * 16 + ln16;       // column N = ln16
        int yl   = pxl >> 6;
        int xl   = pxl & 63;
        int base = ((yl + dy) * 66 + (xl + dx)) * 32 + lhalf * 8;
        bf[ni].u[0] = *(const u32x4*)(lx + base);
        bf[ni].u[1] = *(const u32x4*)(lx + base + 16);
      }
      #pragma unroll
      for (int ni = 0; ni < 4; ++ni) {
        acc[0][ni] = __builtin_amdgcn_wmma_f32_16x16x32_bf16(
            false, a0.v, false, bf[ni].v, (short)0, acc[0][ni], false, false);
        acc[1][ni] = __builtin_amdgcn_wmma_f32_16x16x32_bf16(
            false, a1.v, false, bf[ni].v, (short)0, acc[1][ni], false, false);
      }
    }
  };

#if USE_ASYNC_LDS
  // Zero X halo once: OOB slots are at fixed addresses and are never
  // overwritten by async loads, so they stay zero for the whole kernel.
  {
    u32x4 z = {0u, 0u, 0u, 0u};
    u32x4* lp = (u32x4*)lds_x;
    for (int i = tid; i < (2 * XH) / 8; i += 256) lp[i] = z;
  }
  issueW(0, 0);
  issueX(0, 0);
  for (int cc = 0; cc < 16; ++cc) {
    const int buf = cc & 1;
    __builtin_amdgcn_s_wait_asynccnt(0);   // my chunk-cc loads landed in LDS
    __syncthreads();                       // everyone's landed; prev compute done
    if (cc + 1 < 16) {                     // overlap next chunk with WMMAs
      issueW(cc + 1, buf ^ 1);
      issueX(cc + 1, buf ^ 1);
    }
    computeChunk(buf);
  }
#else
  for (int cc = 0; cc < 16; ++cc) {
    __syncthreads();
    issueW(cc, 0);
    issueX(cc, 0);
    __syncthreads();
    computeChunk(0);
  }
#endif

  // ---- epilogue: demod scale + noise + bias + lrelu*sqrt(2) ----
  const float ns = nstr[0];
  const float g  = 1.41421356237309515f;
  #pragma unroll
  for (int ni = 0; ni < 4; ++ni) {
    int px   = pxBase + wn * 64 + ni * 16 + ln16;        // n = lane%16
    float nz = noise[px] * ns;
    #pragma unroll
    for (int mi = 0; mi < 2; ++mi) {
      int coB = coBase + wm * 32 + mi * 16 + lhalf * 8;  // m = r + 8*(lane/16)
      #pragma unroll
      for (int r = 0; r < 8; ++r) {
        int co  = coB + r;
        float v = acc[mi][ni][r] * dcoefs[b * COUT_ + co] + nz + bias[co];
        v = (v > 0.f ? v : 0.2f * v) * g;
        out[((size_t)b * COUT_ + co) * PX_ + px] = v;
      }
    }
  }
}

// ---------------------------------------------------------------------------
extern "C" void kernel_launch(void* const* d_in, const int* in_sizes, int n_in,
                              void* d_out, int out_size, void* d_ws, size_t ws_size,
                              hipStream_t stream) {
  const float* x    = (const float*)d_in[0];
  const float* w    = (const float*)d_in[1];
  const float* aw   = (const float*)d_in[2];
  const float* ab   = (const float*)d_in[3];
  const float* wt   = (const float*)d_in[4];
  const float* bias = (const float*)d_in[5];
  const float* nc   = (const float*)d_in[6];
  const float* ns   = (const float*)d_in[7];
  float* out = (float*)d_out;

  char* ws = (char*)d_ws;
  float* styles        = (float*)(ws);                         //  32 KB
  float* dcoefs        = (float*)(ws + (32u << 10));           //  32 KB
  float* wsq           = (float*)(ws + (64u << 10));           //   1 MB
  unsigned short* wb   = (unsigned short*)(ws + (64u << 10) + (1u << 20));
  unsigned short* xs   = (unsigned short*)(ws + (64u << 10) + (1u << 20)
                                             + (size_t)9 * COUT_ * CIN_ * 2);

  k_styles<<<(B_ * CIN_ + 255) / 256, 256, 0, stream>>>(w, aw, ab, styles);
  k_wsq<<<(COUT_ * CIN_) / 256, 256, 0, stream>>>(wt, wsq);
  k_dcoef<<<(B_ * COUT_ + 255) / 256, 256, 0, stream>>>(wsq, styles, dcoefs);
  k_packw<<<(9 * COUT_ * CIN_) / 256, 256, 0, stream>>>(wt, wb);

  dim3 gx(PX_ / 16, CIN_ / 16, B_);
  k_packx<<<gx, 256, 0, stream>>>(x, styles, xs);

  dim3 gm(PX_ / 256, COUT_ / 64, B_);
  k_modconv<<<gm, 256, 0, stream>>>(wb, xs, dcoefs, bias, nc, ns, out);
}
